// QloraMLP_3410204033240
// MI455X (gfx1250) — matrix operand order repaired and checked by reading, not compiled, not run on hardware
//
#include <hip/hip_runtime.h>

// Problem dims (fixed by the reference)
#define D_DIM 4096
#define H_DIM 11008
#define T_DIM 8192

// Tiling
#define BM 128   // token rows per block
#define BN 64    // output cols per block
#define KT 32    // K per step (matches WMMA f16 K=32)

typedef _Float16 half_t;
typedef __attribute__((ext_vector_type(16))) _Float16 v16h;
typedef __attribute__((ext_vector_type(8)))  float    v8f;
typedef __attribute__((ext_vector_type(4)))  unsigned int v4u;
typedef __attribute__((ext_vector_type(8)))  int v8i;
typedef __attribute__((ext_vector_type(4)))  int v4i;

#if __has_builtin(__builtin_amdgcn_tensor_load_to_lds) && \
    __has_builtin(__builtin_amdgcn_s_wait_tensorcnt)
#define HAVE_TDM 1
#else
#define HAVE_TDM 0
#endif

__constant__ float NF4_TAB[16] = {
    -1.0f, -0.6961928009986877f, -0.5250730514526367f, -0.39491748809814453f,
    -0.28444138169288635f, -0.18477343022823334f, -0.09105003625154495f, 0.0f,
    0.07958029955625534f, 0.16093020141124725f, 0.24611230194568634f,
    0.33791524171829224f, 0.44070982933044434f, 0.5626170039176941f,
    0.7229568362236023f, 1.0f};

union FragH { v16h v; uint4 q[2]; };
union Pack8H { half_t h[8]; uint4 q; };

// A fragment (16x32 f16, M x K), ISA layout:
// lanes 0-15: M=lane, K in {0..7}U{16..23}; lanes 16-31: M=lane-16, K in {8..15}U{24..31}
static __device__ inline v16h load_a_frag(const half_t* tile, int row_base) {
  const int lane = threadIdx.x & 31;
  const int m    = lane & 15;
  const int koff = (lane >> 4) * 8;
  const half_t* p = tile + (row_base + m) * KT;
  FragH f;
  f.q[0] = *(const uint4*)(p + koff);
  f.q[1] = *(const uint4*)(p + 16 + koff);
  return f.v;
}

// B fragment (32x16 f16, K x N) built from a row-major [N][K] LDS tile (NT gemm):
// lanes 0-15: N=lane, K in {0..7}U{16..23}; lanes 16-31: N=lane-16, K in {8..15}U{24..31}
static __device__ inline v16h load_b_frag(const half_t* tile, int row_base) {
  const int lane = threadIdx.x & 31;
  const int n  = lane & 15;
  const int kb = (lane >> 4) * 8;
  const half_t* p = tile + (row_base + n) * KT + kb;
  FragH f;
  f.q[0] = *(const uint4*)(p);
  f.q[1] = *(const uint4*)(p + 16);
  return f.v;
}

// Dequantize 8 consecutive NF4 codes (int32 each) -> 8 f16 in LDS
static __device__ inline void dequant8(const int* __restrict__ codes, size_t gidx,
                                       float scale, const float* __restrict__ tab,
                                       half_t* __restrict__ dst) {
  int4 ca = *(const int4*)(codes + gidx);
  int4 cb = *(const int4*)(codes + gidx + 4);
  Pack8H o;
  o.h[0] = (half_t)(tab[ca.x & 15] * scale);
  o.h[1] = (half_t)(tab[ca.y & 15] * scale);
  o.h[2] = (half_t)(tab[ca.z & 15] * scale);
  o.h[3] = (half_t)(tab[ca.w & 15] * scale);
  o.h[4] = (half_t)(tab[cb.x & 15] * scale);
  o.h[5] = (half_t)(tab[cb.y & 15] * scale);
  o.h[6] = (half_t)(tab[cb.z & 15] * scale);
  o.h[7] = (half_t)(tab[cb.w & 15] * scale);
  *(uint4*)dst = o.q;
}

#if HAVE_TDM
// Issue a 2D TDM load: tile_d1 x tile_d0 elements (2 bytes each) from a row-major
// tensor (row stride stride_d0 elements) into contiguous LDS at lds_off.
// D# bit layout per CDNA5 ISA 8.3/8.4 (group0: count|lds|global|type, group1:
// data_size|dims|tile|stride). Trailing groups unused for a 2D tile (zeros).
// This toolchain exposes the 6-arg builtin:
//   (uint32x4 g0, int32x8 g1, int32x4, int32x4, int32x8, i32 cpol)
static __device__ inline void tdm_load_2d(unsigned int lds_off, const void* gptr,
                                          unsigned int tensor_d0, unsigned int tensor_d1,
                                          unsigned int tile_d0, unsigned int tile_d1,
                                          unsigned int stride_d0) {
  unsigned long long ga = (unsigned long long)gptr;
  v4u g0;
  g0.x = 0x1u;                                             // count=1 (valid), user D#
  g0.y = lds_off;                                          // lds_addr (bytes)
  g0.z = (unsigned int)ga;                                 // global_addr[31:0]
  g0.w = (unsigned int)((ga >> 32) & 0x1FFFFFFu) | (2u << 30);  // addr[56:32] | type=2
  v8i g1;
  g1[0] = (int)(1u << 16);                                 // data_size=1 -> 2B elements
  g1[1] = (int)((tensor_d0 & 0xFFFFu) << 16);              // tensor_dim0[15:0]
  g1[2] = (int)((tensor_d0 >> 16) | ((tensor_d1 & 0xFFFFu) << 16));
  g1[3] = (int)((tensor_d1 >> 16) | (tile_d0 << 16));      // tensor_dim1 hi | tile_dim0
  g1[4] = (int)(tile_d1 & 0xFFFFu);                        // tile_dim1 (tile_dim2=0)
  g1[5] = (int)stride_d0;                                  // tensor_dim0_stride[31:0]
  g1[6] = 0;                                               // stride hi, dim1_stride lo
  g1[7] = 0;
  v4i z4 = {0, 0, 0, 0};
  v8i z8 = {0, 0, 0, 0, 0, 0, 0, 0};
  __builtin_amdgcn_tensor_load_to_lds(g0, g1, z4, z4, z8, 0);
}
#endif

// ---------------------------------------------------------------------------
// Kernel 0: x fp32 -> fp16 (one shot into workspace)
// ---------------------------------------------------------------------------
__global__ __launch_bounds__(256) void xcvt_f32_to_f16(const float* __restrict__ in,
                                                       half_t* __restrict__ out) {
  size_t i = ((size_t)blockIdx.x * blockDim.x + threadIdx.x) * 8;
  float4 a = *(const float4*)(in + i);
  float4 b = *(const float4*)(in + i + 4);
  Pack8H o;
  o.h[0] = (half_t)a.x; o.h[1] = (half_t)a.y; o.h[2] = (half_t)a.z; o.h[3] = (half_t)a.w;
  o.h[4] = (half_t)b.x; o.h[5] = (half_t)b.y; o.h[6] = (half_t)b.z; o.h[7] = (half_t)b.w;
  *(uint4*)(out + i) = o.q;
}

// ---------------------------------------------------------------------------
// Kernel 1: h = silu(x @ w1^T) * (x @ w2^T), fused NF4 dequant, fp16 output
// grid: (H/BN, T/BM), block: 256 (8 waves; 4x2 wave grid, 32x32 per wave)
// ---------------------------------------------------------------------------
__global__ __launch_bounds__(256) void qlora_gemm1(
    const half_t* __restrict__ x16,
    const int* __restrict__ w1c, const float* __restrict__ w1s,
    const int* __restrict__ w2c, const float* __restrict__ w2s,
    half_t* __restrict__ hout) {
  __shared__ __align__(16) half_t As [BM * KT];
  __shared__ __align__(16) half_t B1s[BN * KT];
  __shared__ __align__(16) half_t B2s[BN * KT];
  __shared__ float tabL[16];

  const int tid = threadIdx.x;
  if (tid < 16) tabL[tid] = NF4_TAB[tid];

  const int n0 = blockIdx.x * BN;
  const int m0 = blockIdx.y * BM;
  const int wave = tid >> 5;
  const int wm = wave >> 1;   // 0..3 (M)
  const int wn = wave & 1;    // 0..1 (N)

  v8f acc1[2][2], acc2[2][2];
  const v8f vzero = {0.f, 0.f, 0.f, 0.f, 0.f, 0.f, 0.f, 0.f};
#pragma unroll
  for (int i = 0; i < 2; ++i)
#pragma unroll
    for (int j = 0; j < 2; ++j) { acc1[i][j] = vzero; acc2[i][j] = vzero; }

  const int wr = tid >> 2;            // 0..63  weight-tile row
  const int wcol = (tid & 3) * 8;     // 0,8,16,24
#if HAVE_TDM
  const unsigned int as_off = (unsigned int)(size_t)(void*)As;
#endif

#pragma unroll 1
  for (int k0 = 0; k0 < D_DIM; k0 += KT) {
    __syncthreads();  // previous iteration's fragment reads done

#if HAVE_TDM
    // wave 0 DMAs the 128x32 f16 x-tile into LDS via the Tensor Data Mover;
    // the DMA overlaps with the NF4 dequant below.
    if (wave == 0)
      tdm_load_2d(as_off, &x16[(size_t)m0 * D_DIM + k0],
                  D_DIM, T_DIM, KT, BM, D_DIM);
#else
    // stage x tile: 128x32 f16 (2 chunks of 8 halves per thread)
#pragma unroll
    for (int it = 0; it < 2; ++it) {
      int c = tid + it * 256;
      int r = c >> 2, cc = (c & 3) * 8;
      *(uint4*)(&As[r * KT + cc]) =
          *(const uint4*)(&x16[(size_t)(m0 + r) * D_DIM + k0 + cc]);
    }
#endif
    // dequantize w1/w2 tiles: 64x32 codes each (8 codes per thread per matrix)
    {
      size_t g    = (size_t)(n0 + wr) * D_DIM + k0 + wcol;
      size_t sidx = (size_t)(n0 + wr) * (D_DIM / 64) + (k0 >> 6);
      dequant8(w1c, g, w1s[sidx], tabL, &B1s[wr * KT + wcol]);
      dequant8(w2c, g, w2s[sidx], tabL, &B2s[wr * KT + wcol]);
      if (k0 + KT < D_DIM) {       // global_prefetch_b8 of next K-step's codes
        __builtin_prefetch(&w1c[g + KT], 0, 1);
        __builtin_prefetch(&w2c[g + KT], 0, 1);
      }
    }
#if HAVE_TDM
    if (wave == 0) __builtin_amdgcn_s_wait_tensorcnt(0);
#endif
    __syncthreads();

    v16h a[2], b1[2], b2[2];
#pragma unroll
    for (int i = 0; i < 2; ++i) a[i] = load_a_frag(As, wm * 32 + i * 16);
#pragma unroll
    for (int j = 0; j < 2; ++j) {
      b1[j] = load_b_frag(B1s, wn * 32 + j * 16);
      b2[j] = load_b_frag(B2s, wn * 32 + j * 16);
    }
#pragma unroll
    for (int i = 0; i < 2; ++i)
#pragma unroll
      for (int j = 0; j < 2; ++j) {
        acc1[i][j] = __builtin_amdgcn_wmma_f32_16x16x32_f16(
            false, a[i], false, b1[j], (short)0, acc1[i][j], false, false);
        acc2[i][j] = __builtin_amdgcn_wmma_f32_16x16x32_f16(
            false, a[i], false, b2[j], (short)0, acc2[i][j], false, false);
      }
  }

  // epilogue: silu gate, fp16 store. C layout: M = r + 8*(lane>>4), N = lane&15
  const int lane = tid & 31;
  const int cn = lane & 15;
  const int rb = (lane >> 4) * 8;
#pragma unroll
  for (int i = 0; i < 2; ++i)
#pragma unroll
    for (int j = 0; j < 2; ++j)
#pragma unroll
      for (int r = 0; r < 8; ++r) {
        float g = acc1[i][j][r];
        float u = acc2[i][j][r];
        float hv = (g / (1.0f + __expf(-g))) * u;
        int row = m0 + wm * 32 + i * 16 + rb + r;
        int col = n0 + wn * 32 + j * 16 + cn;
        hout[(size_t)row * H_DIM + col] = (half_t)hv;
      }
}

// ---------------------------------------------------------------------------
// Kernel 2: out = h @ w3^T, fused NF4 dequant, fp32 output
// grid: (D/BN, T/BM), block 256
// ---------------------------------------------------------------------------
__global__ __launch_bounds__(256) void qlora_gemm2(
    const half_t* __restrict__ h16,
    const int* __restrict__ w3c, const float* __restrict__ w3s,
    float* __restrict__ out) {
  __shared__ __align__(16) half_t As [BM * KT];
  __shared__ __align__(16) half_t B3s[BN * KT];
  __shared__ float tabL[16];

  const int tid = threadIdx.x;
  if (tid < 16) tabL[tid] = NF4_TAB[tid];

  const int n0 = blockIdx.x * BN;   // over D
  const int m0 = blockIdx.y * BM;   // over T
  const int wave = tid >> 5;
  const int wm = wave >> 1;
  const int wn = wave & 1;

  v8f acc[2][2];
  const v8f vzero = {0.f, 0.f, 0.f, 0.f, 0.f, 0.f, 0.f, 0.f};
#pragma unroll
  for (int i = 0; i < 2; ++i)
#pragma unroll
    for (int j = 0; j < 2; ++j) acc[i][j] = vzero;

  const int wr = tid >> 2;
  const int wcol = (tid & 3) * 8;
#if HAVE_TDM
  const unsigned int as_off = (unsigned int)(size_t)(void*)As;
#endif

#pragma unroll 1
  for (int k0 = 0; k0 < H_DIM; k0 += KT) {
    __syncthreads();
#if HAVE_TDM
    if (wave == 0)
      tdm_load_2d(as_off, &h16[(size_t)m0 * H_DIM + k0],
                  H_DIM, T_DIM, KT, BM, H_DIM);
#else
#pragma unroll
    for (int it = 0; it < 2; ++it) {
      int c = tid + it * 256;
      int r = c >> 2, cc = (c & 3) * 8;
      *(uint4*)(&As[r * KT + cc]) =
          *(const uint4*)(&h16[(size_t)(m0 + r) * H_DIM + k0 + cc]);
    }
#endif
    {
      size_t g    = (size_t)(n0 + wr) * H_DIM + k0 + wcol;
      size_t sidx = (size_t)(n0 + wr) * (H_DIM / 64) + (k0 >> 6);
      dequant8(w3c, g, w3s[sidx], tabL, &B3s[wr * KT + wcol]);
      if (k0 + KT < H_DIM) __builtin_prefetch(&w3c[g + KT], 0, 1);
    }
#if HAVE_TDM
    if (wave == 0) __builtin_amdgcn_s_wait_tensorcnt(0);
#endif
    __syncthreads();

    v16h a[2], b[2];
#pragma unroll
    for (int i = 0; i < 2; ++i) a[i] = load_a_frag(As, wm * 32 + i * 16);
#pragma unroll
    for (int j = 0; j < 2; ++j) b[j] = load_b_frag(B3s, wn * 32 + j * 16);
#pragma unroll
    for (int i = 0; i < 2; ++i)
#pragma unroll
      for (int j = 0; j < 2; ++j)
        acc[i][j] = __builtin_amdgcn_wmma_f32_16x16x32_f16(
            false, a[i], false, b[j], (short)0, acc[i][j], false, false);
  }

  const int lane = tid & 31;
  const int cn = lane & 15;
  const int rb = (lane >> 4) * 8;
#pragma unroll
  for (int i = 0; i < 2; ++i)
#pragma unroll
    for (int j = 0; j < 2; ++j)
#pragma unroll
      for (int r = 0; r < 8; ++r) {
        int row = m0 + wm * 32 + i * 16 + rb + r;
        int col = n0 + wn * 32 + j * 16 + cn;
        out[(size_t)row * D_DIM + col] = acc[i][j][r];
      }
}

// ---------------------------------------------------------------------------
extern "C" void kernel_launch(void* const* d_in, const int* in_sizes, int n_in,
                              void* d_out, int out_size, void* d_ws, size_t ws_size,
                              hipStream_t stream) {
  (void)in_sizes; (void)n_in; (void)out_size; (void)ws_size;
  const float* x   = (const float*)d_in[0];
  const int*   w1c = (const int*)  d_in[1];
  const float* w1s = (const float*)d_in[2];
  const int*   w2c = (const int*)  d_in[3];
  const float* w2s = (const float*)d_in[4];
  const int*   w3c = (const int*)  d_in[5];
  const float* w3s = (const float*)d_in[6];
  float* out = (float*)d_out;

  // workspace: [x16: T*D f16][h16: T*H f16]
  half_t* x16 = (half_t*)d_ws;
  half_t* h16 = (half_t*)((char*)d_ws + (size_t)T_DIM * D_DIM * sizeof(half_t));

  const size_t nx = (size_t)T_DIM * D_DIM;
  xcvt_f32_to_f16<<<dim3((unsigned)(nx / (256 * 8))), dim3(256), 0, stream>>>(x, x16);

  qlora_gemm1<<<dim3(H_DIM / BN, T_DIM / BM), dim3(256), 0, stream>>>(
      x16, w1c, w1s, w2c, w2s, h16);

  qlora_gemm2<<<dim3(D_DIM / BN, T_DIM / BM), dim3(256), 0, stream>>>(
      h16, w3c, w3s, out);
}
